// TreeNet_73315091743420
// MI455X (gfx1250) — matrix operand order repaired and checked by reading, not compiled, not run on hardware
//
#include <hip/hip_runtime.h>
#include <hip/hip_bf16.h>

// ---------------------------------------------------------------------------
// Tree-LSTM bottom-up sweep for gfx1250 (MI455X), wave32 + WMMA f16->f32.
// M=32 rows per workgroup: every B fragment (32x16 f16 weight tile) feeds two
// independent WMMAs (two A tiles), halving L2 weight traffic and hiding the
// B-load wait. Column-sliced gate ownership keeps all gate nonlinearity
// register-resident in the WMMA D layout (VGPR i -> row i+8*(lane>>4),
// col n0+(lane&15)); only h_left round-trips through LDS.
// h state stored f16, c state stored f32 (recurrent precision path).
// ---------------------------------------------------------------------------

typedef __attribute__((ext_vector_type(16))) _Float16 v16h;
typedef __attribute__((ext_vector_type(8)))  _Float16 v8h;
typedef __attribute__((ext_vector_type(8)))  float    v8f;

#define DH     256
#define NG     768
#define KPAD   320
#define LEAVES 16384

__device__ __forceinline__ float sigmoidf_(float x) { return 1.0f / (1.0f + __expf(-x)); }

__device__ __forceinline__ v8f wmma16(v16h a, v16h b, v8f c) {
  return __builtin_amdgcn_wmma_f32_16x16x32_f16(false, a, false, b, (short)0, c, false, false);
}

// A fragment (16x32 f16) from a row-major f16 row pointer (this lane's row).
// Lanes 0-15 -> K = k0+0..7 , k0+16..23 ; lanes 16-31 -> K = k0+8..15 , k0+24..31.
__device__ __forceinline__ v16h load_a_f16(const _Float16* rowptr, int k0, int half) {
  const v8h* p0 = (const v8h*)(rowptr + k0 + half * 8);
  const v8h* p1 = (const v8h*)(rowptr + k0 + 16 + half * 8);
  v8h lo = *p0, hi = *p1;
  v16h a;
#pragma unroll
  for (int i = 0; i < 8; i++) { a[i] = lo[i]; a[8 + i] = hi[i]; }
  return a;
}

// B fragment (32x16 f16) from [N][K]-transposed f16 weights.
// Lanes 0-15 hold K = k0+0..7 , k0+16..23 of column n0+(lane&15), lanes 16-31 hold K = k0+8..15 , k0+24..31.
__device__ __forceinline__ v16h load_b_t(const _Float16* WT, int ldk, int n0, int k0, int lane) {
  int n  = n0 + (lane & 15);
  int kb = k0 + ((lane >> 4) << 3);
  const v8h* p = (const v8h*)(WT + (size_t)n * ldk + kb);
  v8h lo = p[0], hi = p[2];
  v16h b;
#pragma unroll
  for (int i = 0; i < 8; i++) { b[i] = lo[i]; b[8 + i] = hi[i]; }
  return b;
}

// ---------------------------------------------------------------------------
// prep: f32 weights -> f16 [N][K] transposed; bsum = bs + bc
// ---------------------------------------------------------------------------
__global__ __launch_bounds__(256) void prep_kernel(
    const float* __restrict__ Wx, const float* __restrict__ Ws, const float* __restrict__ Wc,
    const float* __restrict__ bs, const float* __restrict__ bc,
    _Float16* __restrict__ WxT, _Float16* __restrict__ WcT, _Float16* __restrict__ WsT,
    float* __restrict__ bsum) {
  int i = blockIdx.x * 256 + threadIdx.x;
  if (i < NG * KPAD) {
    int n = i / KPAD, k = i % KPAD;
    WxT[i] = (k < 300) ? (_Float16)Wx[(size_t)k * NG + n] : (_Float16)0.0f;
  }
  int j = i - NG * KPAD;
  if (j >= 0 && j < NG * DH) {
    int n = j / DH, k = j % DH;
    WcT[j] = (_Float16)Wc[(size_t)k * NG + n];
    WsT[j] = (_Float16)Ws[(size_t)k * NG + n];
  }
  int b = j - NG * DH;
  if (b >= 0 && b < NG) bsum[b] = bs[b] + bc[b];
}

// ---------------------------------------------------------------------------
// gather: X[leaf][k] = f16(emb[word_ids[leaf]][k]), K padded 300 -> 320
// ---------------------------------------------------------------------------
__global__ __launch_bounds__(256) void gather_kernel(
    const int* __restrict__ word_ids, const float* __restrict__ emb,
    _Float16* __restrict__ X) {
  int idx = blockIdx.x * 256 + threadIdx.x;   // LEAVES*KPAD total
  int leaf = idx / KPAD, k = idx % KPAD;
  float v = (k < 300) ? emb[(size_t)word_ids[leaf] * 300 + k] : 0.0f;
  X[idx] = (_Float16)v;
}

// ---------------------------------------------------------------------------
// leaf: 32 leaves/WG; G = X @ Wx (K=320); gates gi|go|cpre register-resident
// ---------------------------------------------------------------------------
__global__ __launch_bounds__(256) void leaf_kernel(
    const _Float16* __restrict__ X, const _Float16* __restrict__ WxT,
    const float* __restrict__ bx,
    _Float16* __restrict__ h_out, float* __restrict__ c_out) {
  int tid = threadIdx.x, lane = tid & 31, wave = tid >> 5, half = lane >> 4;
  int l0 = blockIdx.x * 32;
  const _Float16* arow0 = X + (size_t)(l0 + (lane & 15)) * KPAD;
  const _Float16* arow1 = X + (size_t)(l0 + 16 + (lane & 15)) * KPAD;

  v8f aGI[2][2] = {}, aGO[2][2] = {}, aCP[2][2] = {};
  for (int k0 = 0; k0 < KPAD; k0 += 32) {
    v16h a0 = load_a_f16(arow0, k0, half);
    v16h a1 = load_a_f16(arow1, k0, half);
#pragma unroll
    for (int t = 0; t < 2; t++) {
      int n0 = wave * 32 + t * 16;
      v16h b;
      b = load_b_t(WxT, KPAD, n0,       k0, lane);
      aGI[0][t] = wmma16(a0, b, aGI[0][t]);
      aGI[1][t] = wmma16(a1, b, aGI[1][t]);
      b = load_b_t(WxT, KPAD, 256 + n0, k0, lane);
      aGO[0][t] = wmma16(a0, b, aGO[0][t]);
      aGO[1][t] = wmma16(a1, b, aGO[1][t]);
      b = load_b_t(WxT, KPAD, 512 + n0, k0, lane);
      aCP[0][t] = wmma16(a0, b, aCP[0][t]);
      aCP[1][t] = wmma16(a1, b, aCP[1][t]);
    }
  }

  int rb = half * 8, nc = lane & 15;
#pragma unroll
  for (int m = 0; m < 2; m++) {
#pragma unroll
    for (int t = 0; t < 2; t++) {
#pragma unroll
      for (int i = 0; i < 8; i++) {
        int col = wave * 32 + t * 16 + nc;
        int row = l0 + m * 16 + rb + i;
        float gi = aGI[m][t][i] + bx[col];
        float go = aGO[m][t][i] + bx[256 + col];
        float cp = aCP[m][t][i] + bx[512 + col];
        float c = sigmoidf_(gi) * tanhf(cp);
        float h = sigmoidf_(go) * tanhf(c);
        c_out[(size_t)row * DH + col] = c;
        h_out[(size_t)row * DH + col] = (_Float16)h;
      }
    }
  }
}

// ---------------------------------------------------------------------------
// combine level: 32 parents/WG, fused left-child + parent GEMMs, one barrier
// ---------------------------------------------------------------------------
__global__ __launch_bounds__(256) void combine_kernel(
    const _Float16* __restrict__ h_in, const float* __restrict__ c_in,
    _Float16* __restrict__ h_out, float* __restrict__ c_out,
    const _Float16* __restrict__ WcT, const _Float16* __restrict__ WsT,
    const float* __restrict__ bsum, int P) {
  __shared__ _Float16 hl[32 * DH];

  int tid = threadIdx.x, lane = tid & 31, wave = tid >> 5, half = lane >> 4;
  int p0 = blockIdx.x * 32;
  int pr0 = p0 + (lane & 15);      if (pr0 >= P) pr0 = P - 1;  // clamp: EXEC stays all-ones
  int pr1 = p0 + 16 + (lane & 15); if (pr1 >= P) pr1 = P - 1;

  // ---- phase 1: G_l = H_even @ Wc (gc/go columns of this wave's slice) ----
  const _Float16* arowE0 = h_in + (size_t)(2 * pr0) * DH;
  const _Float16* arowE1 = h_in + (size_t)(2 * pr1) * DH;
  v8f gcl[2][2] = {}, gol[2][2] = {};
  for (int k0 = 0; k0 < DH; k0 += 32) {
    v16h a0 = load_a_f16(arowE0, k0, half);
    v16h a1 = load_a_f16(arowE1, k0, half);
#pragma unroll
    for (int t = 0; t < 2; t++) {
      int n0 = wave * 32 + t * 16;
      v16h b;
      b = load_b_t(WcT, DH, 256 + n0, k0, lane);
      gcl[0][t] = wmma16(a0, b, gcl[0][t]);
      gcl[1][t] = wmma16(a1, b, gcl[1][t]);
      b = load_b_t(WcT, DH, 512 + n0, k0, lane);
      gol[0][t] = wmma16(a0, b, gol[0][t]);
      gol[1][t] = wmma16(a1, b, gol[1][t]);
    }
  }

  // ---- left-child nonlinearity: c_l in regs (phase-3 layout), h_l -> LDS ----
  int rb = half * 8, nc = lane & 15;
  float cl[2][2][8];
#pragma unroll
  for (int m = 0; m < 2; m++) {
#pragma unroll
    for (int t = 0; t < 2; t++) {
#pragma unroll
      for (int i = 0; i < 8; i++) {
        int col = wave * 32 + t * 16 + nc;
        int pri = p0 + m * 16 + rb + i;
        if (pri >= P) pri = P - 1;
        float gc = gcl[m][t][i] + bsum[256 + col];
        float go = gol[m][t][i] + bsum[512 + col];
        float clv = sigmoidf_(gc) * c_in[(size_t)(2 * pri) * DH + col];
        cl[m][t][i] = clv;
        hl[(m * 16 + rb + i) * DH + col] = (_Float16)(sigmoidf_(go) * tanhf(clv));
      }
    }
  }
  __syncthreads();

  // ---- phase 2: G_p = H_odd @ Wc + h_l @ Ws (full gate slice) ----
  const _Float16* arowO0 = h_in + (size_t)(2 * pr0 + 1) * DH;
  const _Float16* arowO1 = h_in + (size_t)(2 * pr1 + 1) * DH;
  const _Float16* arowL0 = hl + (size_t)(lane & 15) * DH;
  const _Float16* arowL1 = hl + (size_t)(16 + (lane & 15)) * DH;
  v8f gs2[2][2] = {}, gc2[2][2] = {}, go2[2][2] = {};
  for (int k0 = 0; k0 < DH; k0 += 32) {
    v16h aO0 = load_a_f16(arowO0, k0, half);
    v16h aO1 = load_a_f16(arowO1, k0, half);
    v16h aL0 = load_a_f16(arowL0, k0, half);  // ds_load_b128
    v16h aL1 = load_a_f16(arowL1, k0, half);
#pragma unroll
    for (int t = 0; t < 2; t++) {
      int n0 = wave * 32 + t * 16;
      v16h b;
      b = load_b_t(WcT, DH, n0, k0, lane);
      gs2[0][t] = wmma16(aO0, b, gs2[0][t]);
      gs2[1][t] = wmma16(aO1, b, gs2[1][t]);
      b = load_b_t(WsT, DH, n0, k0, lane);
      gs2[0][t] = wmma16(aL0, b, gs2[0][t]);
      gs2[1][t] = wmma16(aL1, b, gs2[1][t]);
      b = load_b_t(WcT, DH, 256 + n0, k0, lane);
      gc2[0][t] = wmma16(aO0, b, gc2[0][t]);
      gc2[1][t] = wmma16(aO1, b, gc2[1][t]);
      b = load_b_t(WsT, DH, 256 + n0, k0, lane);
      gc2[0][t] = wmma16(aL0, b, gc2[0][t]);
      gc2[1][t] = wmma16(aL1, b, gc2[1][t]);
      b = load_b_t(WcT, DH, 512 + n0, k0, lane);
      go2[0][t] = wmma16(aO0, b, go2[0][t]);
      go2[1][t] = wmma16(aO1, b, go2[1][t]);
      b = load_b_t(WsT, DH, 512 + n0, k0, lane);
      go2[0][t] = wmma16(aL0, b, go2[0][t]);
      go2[1][t] = wmma16(aL1, b, go2[1][t]);
    }
  }

  // ---- parent nonlinearity + writeback (register-resident) ----
#pragma unroll
  for (int m = 0; m < 2; m++) {
#pragma unroll
    for (int t = 0; t < 2; t++) {
#pragma unroll
      for (int i = 0; i < 8; i++) {
        int prow = p0 + m * 16 + rb + i;
        if (prow < P) {
          int col = wave * 32 + t * 16 + nc;
          float gs = gs2[m][t][i] + bsum[col];
          float gc = gc2[m][t][i] + bsum[256 + col];
          float go = go2[m][t][i] + bsum[512 + col];
          float cp = sigmoidf_(gc) * c_in[(size_t)(2 * prow + 1) * DH + col] +
                     sigmoidf_(gs) * cl[m][t][i];
          float hp = sigmoidf_(go) * tanhf(cp);
          c_out[(size_t)prow * DH + col] = cp;
          h_out[(size_t)prow * DH + col] = (_Float16)hp;
        }
      }
    }
  }
}

// ---------------------------------------------------------------------------
// root: single node, sibling = zeros; only gc/go needed -> 512-thread GEMV
// ---------------------------------------------------------------------------
__global__ __launch_bounds__(512) void root_kernel(
    const _Float16* __restrict__ h, const float* __restrict__ c,
    const _Float16* __restrict__ WcT, const float* __restrict__ bsum,
    float* __restrict__ out) {
  __shared__ float sg[512];
  int j = threadIdx.x;
  int col = 256 + j;                       // j<256 -> gc, j>=256 -> go
  float acc = bsum[col];
  const _Float16* w = WcT + (size_t)col * DH;
  for (int k = 0; k < DH; k++) acc += (float)h[k] * (float)w[k];
  sg[j] = sigmoidf_(acc);
  __syncthreads();
  if (j < 256) {
    float cn = sg[j] * c[j];
    out[j] = sg[256 + j] * tanhf(cn);
  }
}

// ---------------------------------------------------------------------------
extern "C" void kernel_launch(void* const* d_in, const int* in_sizes, int n_in,
                              void* d_out, int out_size, void* d_ws, size_t ws_size,
                              hipStream_t stream) {
  const int*   word_ids = (const int*)d_in[0];
  const float* emb      = (const float*)d_in[1];
  const float* Wx       = (const float*)d_in[2];
  const float* bx       = (const float*)d_in[3];
  const float* Ws       = (const float*)d_in[4];
  const float* bs       = (const float*)d_in[5];
  const float* Wc       = (const float*)d_in[6];
  const float* bc       = (const float*)d_in[7];
  float* out = (float*)d_out;

  size_t off = 0;
  char* base = (char*)d_ws;
  auto alloc = [&](size_t bytes) -> char* {
    char* p = base + off;
    off += (bytes + 255) & ~(size_t)255;
    return p;
  };
  _Float16* WxT  = (_Float16*)alloc((size_t)NG * KPAD * 2);
  _Float16* WcT  = (_Float16*)alloc((size_t)NG * DH * 2);
  _Float16* WsT  = (_Float16*)alloc((size_t)NG * DH * 2);
  float*    bsum = (float*)alloc((size_t)NG * 4);
  _Float16* X    = (_Float16*)alloc((size_t)LEAVES * KPAD * 2);
  _Float16* hA   = (_Float16*)alloc((size_t)LEAVES * DH * 2);
  _Float16* hB   = (_Float16*)alloc((size_t)LEAVES * DH * 2);
  float*    cA   = (float*)alloc((size_t)LEAVES * DH * 4);
  float*    cB   = (float*)alloc((size_t)LEAVES * DH * 4);

  int prep_elems = NG * KPAD + NG * DH + NG;
  prep_kernel<<<(prep_elems + 255) / 256, 256, 0, stream>>>(
      Wx, Ws, Wc, bs, bc, WxT, WcT, WsT, bsum);

  gather_kernel<<<(LEAVES * KPAD) / 256, 256, 0, stream>>>(word_ids, emb, X);

  leaf_kernel<<<LEAVES / 32, 256, 0, stream>>>(X, WxT, bx, hA, cA);

  _Float16* hin = hA; float* cin = cA;
  _Float16* hout = hB; float* cout = cB;
  for (int lvl = 0; lvl < 14; lvl++) {
    int P = LEAVES >> (lvl + 1);
    int grid = (P + 31) / 32;
    combine_kernel<<<grid, 256, 0, stream>>>(hin, cin, hout, cout, WcT, WsT, bsum, P);
    _Float16* th = hin; hin = hout; hout = th;
    float* tc = cin; cin = cout; cout = tc;
  }

  root_kernel<<<1, 512, 0, stream>>>(hin, cin, WcT, bsum, out);
}
